// SelfAttentionBlock_56470230008346
// MI455X (gfx1250) — matrix operand order repaired and checked by reading, not compiled, not run on hardware
//
#include <hip/hip_runtime.h>
#include <stdint.h>

// Problem constants (match reference)
#define B_    4
#define N_    1024
#define D_    1024
#define H_    16
#define DH_   64
#define HID_  2730
#define HIDP_ 2752          // HID padded to multiple of 64 (zero-padded weights)
#define SCALE_ 0.125f       // DH^-0.5
#define NEG_  -1e9f
#define EPS_  1e-6f

#if defined(__has_builtin)
#if __has_builtin(__builtin_amdgcn_tensor_load_to_lds) && __has_builtin(__builtin_amdgcn_s_wait_tensorcnt)
#define USE_TDM 1
#endif
#endif
#ifndef USE_TDM
#define USE_TDM 0
#endif

typedef __attribute__((ext_vector_type(16))) __bf16 bf16x16;
typedef __attribute__((ext_vector_type(8)))  float  floatx8;

union FragAB { uint4 u4[2]; bf16x16 v; };

__device__ __forceinline__ unsigned short f2bf(float f) {
  unsigned int u = __float_as_uint(f);
  unsigned int r = u + 0x7FFFu + ((u >> 16) & 1u);   // round-to-nearest-even
  return (unsigned short)(r >> 16);
}
__device__ __forceinline__ float bf2f(unsigned short h) {
  return __uint_as_float(((unsigned int)h) << 16);
}

#if USE_TDM
typedef unsigned int u32x4 __attribute__((ext_vector_type(4)));
typedef int          i32x8 __attribute__((ext_vector_type(8)));
typedef int          i32x4 __attribute__((ext_vector_type(4)));

// Issue one TDM 2D tile load: (tileRows x 32) bf16 tile from row-major global
// (leading dim `ld` elements) into LDS at byte offset ldsOff, padding the LDS
// destination by 16B after every 64B so rows land at the KP=40 element stride.
// rowsLeft/colsLeft are the remaining tensor extents from the tile origin:
// TDM zero-fills OOB rows, which replaces the edge-tile guards.
__device__ __forceinline__ void tdm_load_tile(const unsigned short* gsrc, int ld,
                                              int rowsLeft, int colsLeft,
                                              unsigned ldsOff, int tileRows)
{
  unsigned long long ga = (unsigned long long)(uintptr_t)gsrc;
  u32x4 g0;
  g0.x = 1u;                                                   // count=1 (valid)
  g0.y = ldsOff;                                               // lds_addr
  g0.z = (unsigned)(ga & 0xFFFFFFFFull);                       // global_addr lo
  g0.w = (unsigned)((ga >> 32) & 0x01FFFFFFull) | (2u << 30);  // addr hi | type=2
  const unsigned d0 = (unsigned)colsLeft;                      // tensor_dim0
  const unsigned d1 = (unsigned)rowsLeft;                      // tensor_dim1
  i32x8 g1;
  g1[0] = (int)((1u << 16)      // data_size = 2 bytes
              | (1u << 20)      // pad_enable
              | (3u << 22)      // pad_interval: every 16 DWORDs (64B row)
              | (3u << 25));    // pad_amount: 4 DWORDs (8 bf16) -> KP=40
  g1[1] = (int)((d0 & 0xFFFFu) << 16);                         // dim0[15:0]
  g1[2] = (int)((d0 >> 16) | ((d1 & 0xFFFFu) << 16));          // dim0[31:16] dim1[15:0]
  g1[3] = (int)((d1 >> 16) | (32u << 16));                     // dim1[31:16] tile_dim0=32
  g1[4] = (int)((unsigned)tileRows & 0xFFFFu);                 // tile_dim1 (tile_dim2=0)
  g1[5] = (int)(unsigned)ld;                                   // tensor_dim0_stride
  g1[6] = 0;
  g1[7] = 0;
  i32x4 gz4 = {0, 0, 0, 0};
  i32x8 gz8 = {0, 0, 0, 0, 0, 0, 0, 0};
  // amdgpu-toolchain (clang-23 / therock-10.0) 6-arg form
  __builtin_amdgcn_tensor_load_to_lds(g0, g1, gz4, gz4, gz8, 0);
}
#endif

// ---------------------------------------------------------------------------
// Tiled bf16 WMMA GEMM:  C[M][N] = A[M][K] (row major, bf16) @ Bt[N][K]^T + bias
// Block tile 256x128, 8 wave32s, each wave owns a 64x64 patch (16 WMMAs/K-step).
// Global->LDS copies via Tensor Data Mover (double-buffered), LDS rows padded
// to 40 elements.  EPI 0: f32 (+bias)  EPI 1: bf16 (+bias)
// EPI 2: f32, C = (acc+bias)*ls[col] + res[row][col]
// blockIdx.z batches A/B/C by (sAz, sBz, sCz) elements (attention heads).
// ---------------------------------------------------------------------------
template <int EPI>
__global__ __launch_bounds__(256)
void gemm_bf16_wmma(const unsigned short* __restrict__ A, long long sAz, int lda,
                    const unsigned short* __restrict__ Bt, long long sBz, int ldb,
                    void* __restrict__ Cv, long long sCz, int ldc,
                    const float* __restrict__ bias,
                    const float* __restrict__ ls,
                    const float* __restrict__ res, int ldres,
                    int M, int N, int K)
{
  constexpr int BM = 256, BN = 128, KT = 32, KP = 40;
  __shared__ __align__(16) unsigned short As[2][BM * KP];   // 40 KB
  __shared__ __align__(16) unsigned short Bs[2][BN * KP];   // 20 KB

  const int z = blockIdx.z;
  A  += (long long)z * sAz;
  Bt += (long long)z * sBz;
  float*          Cf = (float*)Cv          + (long long)z * sCz;
  unsigned short* Ch = (unsigned short*)Cv + (long long)z * sCz;

  const int tid  = threadIdx.x;
  const int lane = tid & 31;
  const int wave = tid >> 5;
  const int hf   = lane >> 4;     // half-wave select (ISA 16-bit operand layout)
  const int l16  = lane & 15;
  const int wy   = wave >> 1;     // 4x2 wave grid, 64x64 patch per wave
  const int wx   = wave & 1;
  const int rowBase = blockIdx.y * BM;
  const int colBase = blockIdx.x * BN;

  floatx8 acc[4][4];
  {
    floatx8 zf = {};
#pragma unroll
    for (int i = 0; i < 4; ++i)
#pragma unroll
      for (int j = 0; j < 4; ++j) acc[i][j] = zf;
  }

  auto compute = [&](int stage) {
    FragAB bfr[4];
#pragma unroll
    for (int nt = 0; nt < 4; ++nt) {
      const unsigned short* bb = &Bs[stage][(wx * 64 + nt * 16 + l16) * KP];
      bfr[nt].u4[0] = *(const uint4*)(bb + hf * 8);
      bfr[nt].u4[1] = *(const uint4*)(bb + 16 + hf * 8);
    }
#pragma unroll
    for (int mt = 0; mt < 4; ++mt) {
      FragAB af;
      const unsigned short* ab = &As[stage][(wy * 64 + mt * 16 + l16) * KP];
      af.u4[0] = *(const uint4*)(ab + hf * 8);
      af.u4[1] = *(const uint4*)(ab + 16 + hf * 8);
#pragma unroll
      for (int nt = 0; nt < 4; ++nt) {
        acc[mt][nt] = __builtin_amdgcn_wmma_f32_16x16x32_bf16(
            false, af.v, false, bfr[nt].v, (short)0, acc[mt][nt], false, false);
      }
    }
  };

  const int nk = K / KT;     // K is always a multiple of 32 here

#if USE_TDM
  const unsigned ldsA0 = (unsigned)(uintptr_t)&As[0][0];
  const unsigned ldsA1 = (unsigned)(uintptr_t)&As[1][0];
  const unsigned ldsB0 = (unsigned)(uintptr_t)&Bs[0][0];
  const unsigned ldsB1 = (unsigned)(uintptr_t)&Bs[1][0];
  auto issue = [&](int kt, int stage) {
    tdm_load_tile(A + (long long)rowBase * lda + kt, lda,
                  M - rowBase, K - kt, stage ? ldsA1 : ldsA0, BM);
    tdm_load_tile(Bt + (long long)colBase * ldb + kt, ldb,
                  N - colBase, K - kt, stage ? ldsB1 : ldsB0, BN);
  };
  if (wave == 0) {
    issue(0, 0);
    __builtin_amdgcn_s_wait_tensorcnt(0);
  }
  __syncthreads();
  for (int t = 0; t < nk; ++t) {
    if (t + 1 < nk && wave == 0) issue((t + 1) * KT, (t + 1) & 1);  // overlap DMA
    compute(t & 1);
    if (t + 1 < nk) {
      if (wave == 0) __builtin_amdgcn_s_wait_tensorcnt(0);
      __syncthreads();
    }
  }
#else
  // Fallback: global->VGPR->LDS double-buffered copy
  uint4 ra[4], rb[2];
  auto gload = [&](int kt) {
#pragma unroll
    for (int i = 0; i < 4; ++i) {         // A: 1024 chunks of 8 bf16
      int chunk = tid + 256 * i;
      int r = chunk >> 2, c = (chunk & 3) * 8;
      int gr = rowBase + r;
      uint4 v = make_uint4(0u, 0u, 0u, 0u);
      if (gr < M) v = *(const uint4*)(A + (long long)gr * lda + kt + c);
      ra[i] = v;
    }
#pragma unroll
    for (int i = 0; i < 2; ++i) {         // B: 512 chunks
      int chunk = tid + 256 * i;
      int r = chunk >> 2, c = (chunk & 3) * 8;
      int gr = colBase + r;
      uint4 v = make_uint4(0u, 0u, 0u, 0u);
      if (gr < N) v = *(const uint4*)(Bt + (long long)gr * ldb + kt + c);
      rb[i] = v;
    }
  };
  auto sstore = [&](int stage) {
#pragma unroll
    for (int i = 0; i < 4; ++i) {
      int chunk = tid + 256 * i;
      int r = chunk >> 2, c = (chunk & 3) * 8;
      *(uint4*)&As[stage][r * KP + c] = ra[i];
    }
#pragma unroll
    for (int i = 0; i < 2; ++i) {
      int chunk = tid + 256 * i;
      int r = chunk >> 2, c = (chunk & 3) * 8;
      *(uint4*)&Bs[stage][r * KP + c] = rb[i];
    }
  };
  gload(0); sstore(0);
  __syncthreads();
  for (int t = 0; t < nk; ++t) {
    if (t + 1 < nk) gload((t + 1) * KT);
    compute(t & 1);
    if (t + 1 < nk) {
      __syncthreads();
      sstore((t + 1) & 1);
      __syncthreads();
    }
  }
#endif

  // Epilogue.  C/D layout: VGPR j, lanes<16 -> row j, lanes>=16 -> row j+8; col=l16.
#pragma unroll
  for (int mt = 0; mt < 4; ++mt) {
#pragma unroll
    for (int nt = 0; nt < 4; ++nt) {
      const int r0 = rowBase + wy * 64 + mt * 16 + hf * 8;
      const int c  = colBase + wx * 64 + nt * 16 + l16;
      if (c >= N) continue;
      const float bi  = bias ? bias[c] : 0.f;
      const float lsv = (EPI == 2) ? ls[c] : 0.f;
#pragma unroll
      for (int j = 0; j < 8; ++j) {
        const int r = r0 + j;
        if (r >= M) continue;
        float v = acc[mt][nt][j] + bi;
        const long long idx = (long long)r * ldc + c;
        if (EPI == 0)      Cf[idx] = v;
        else if (EPI == 1) Ch[idx] = f2bf(v);
        else               Cf[idx] = v * lsv + res[(long long)r * ldres + c];
      }
    }
  }
}

// ---------------------------------------------------------------------------
// LayerNorm over D=1024, f32 in -> bf16 out.  One block (256 thr) per row.
// ---------------------------------------------------------------------------
__global__ __launch_bounds__(256)
void layernorm_bf16_kernel(const float* __restrict__ x, const float* __restrict__ g,
                           const float* __restrict__ b, unsigned short* __restrict__ out)
{
  __shared__ float red[256];
  const int row = blockIdx.x;
  const int tid = threadIdx.x;
  const float* xr = x + (long long)row * D_;
  float v[4], s = 0.f;
#pragma unroll
  for (int i = 0; i < 4; ++i) { v[i] = xr[tid + 256 * i]; s += v[i]; }
  red[tid] = s; __syncthreads();
  for (int st = 128; st > 0; st >>= 1) { if (tid < st) red[tid] += red[tid + st]; __syncthreads(); }
  const float mean = red[0] * (1.f / D_);
  __syncthreads();
  float q = 0.f;
#pragma unroll
  for (int i = 0; i < 4; ++i) { float d = v[i] - mean; q += d * d; }
  red[tid] = q; __syncthreads();
  for (int st = 128; st > 0; st >>= 1) { if (tid < st) red[tid] += red[tid + st]; __syncthreads(); }
  const float rs = rsqrtf(red[0] * (1.f / D_) + EPS_);
  unsigned short* orow = out + (long long)row * D_;
#pragma unroll
  for (int i = 0; i < 4; ++i) {
    int c = tid + 256 * i;
    orow[c] = f2bf((v[i] - mean) * rs * g[c] + b[c]);
  }
}

// ---------------------------------------------------------------------------
// Split fused QKV (bf16 [B*N][3D]) into Q (scaled), K as (B,H,N,DH) and
// V transposed to (B,H,DH,N) so the P@V GEMM sees Bt=[N][K].
// ---------------------------------------------------------------------------
__global__ __launch_bounds__(256)
void split_qkv_kernel(const unsigned short* __restrict__ qkv,
                      unsigned short* __restrict__ q,
                      unsigned short* __restrict__ k,
                      unsigned short* __restrict__ vt)
{
  long long idx = (long long)blockIdx.x * 256 + threadIdx.x;   // < B*N*D (exact grid)
  const int c = (int)(idx & 1023);
  const int n = (int)((idx >> 10) & 1023);
  const int b = (int)(idx >> 20);
  const int h = c >> 6, d = c & 63;
  const unsigned short* base = qkv + ((long long)(b * N_ + n)) * (3 * D_);
  const long long qi = (((long long)(b * H_ + h) * N_) + n) * DH_ + d;
  const long long vi = (((long long)(b * H_ + h) * DH_) + d) * N_ + n;
  q[qi]  = f2bf(bf2f(base[c]) * SCALE_);
  k[qi]  = base[D_ + c];
  vt[vi] = base[2 * D_ + c];
}

// ---------------------------------------------------------------------------
// Talking-heads mix (pre), masked softmax over keys, mix (post).
// One block per query row q; all 16 heads resident in LDS (bf16, 33KB).
// Input: per-batch scores f32 [H][N][N]; output: bf16 probs [H][N][N].
// ---------------------------------------------------------------------------
__global__ __launch_bounds__(256)
void mix_softmax_kernel(const float* __restrict__ S, const unsigned char* __restrict__ mask,
                        const float* __restrict__ projl, const float* __restrict__ plb,
                        const float* __restrict__ projw, const float* __restrict__ pwb,
                        unsigned short* __restrict__ P)
{
  __shared__ unsigned short sbuf[H_][N_ + 8];
  __shared__ float wl[H_ * H_], ww[H_ * H_];
  __shared__ float mrow[H_], zrow[H_];
  const int q = blockIdx.x;
  const int tid = threadIdx.x;
  wl[tid] = projl[tid];               // 256 == H*H
  ww[tid] = projw[tid];
  for (int h = 0; h < H_; ++h) {
    const float* sr = S + ((long long)h * N_ + q) * N_;
    for (int c = tid; c < N_; c += 256) {
      float s = sr[c];
      if (!mask[c]) s += NEG_;
      sbuf[h][c] = f2bf(s);
    }
  }
  __syncthreads();
  // pre-softmax head mix (in place, one column per thread at a time)
  for (int c = tid; c < N_; c += 256) {
    float v[H_], o[H_];
#pragma unroll
    for (int h = 0; h < H_; ++h) v[h] = bf2f(sbuf[h][c]);
#pragma unroll
    for (int g = 0; g < H_; ++g) {
      float a = plb[g];
#pragma unroll
      for (int h = 0; h < H_; ++h) a += v[h] * wl[h * H_ + g];
      o[g] = a;
    }
#pragma unroll
    for (int g = 0; g < H_; ++g) sbuf[g][c] = f2bf(o[g]);
  }
  __syncthreads();
  // per-head row max / sum(exp): each wave32 owns two heads, shuffle-reduce
  const int wave = tid >> 5, lane = tid & 31;
#pragma unroll
  for (int gi = 0; gi < 2; ++gi) {
    const int g = wave * 2 + gi;
    float mx = -3.0e38f;
    for (int c = lane; c < N_; c += 32) mx = fmaxf(mx, bf2f(sbuf[g][c]));
    for (int off = 16; off > 0; off >>= 1) mx = fmaxf(mx, __shfl_xor(mx, off, 32));
    float sm = 0.f;
    for (int c = lane; c < N_; c += 32) sm += __expf(bf2f(sbuf[g][c]) - mx);
    for (int off = 16; off > 0; off >>= 1) sm += __shfl_xor(sm, off, 32);
    if (lane == 0) { mrow[g] = mx; zrow[g] = 1.f / sm; }
  }
  __syncthreads();
  // normalize, post-softmax head mix, store bf16 probabilities
  for (int c = tid; c < N_; c += 256) {
    float pv[H_];
#pragma unroll
    for (int h = 0; h < H_; ++h) pv[h] = __expf(bf2f(sbuf[h][c]) - mrow[h]) * zrow[h];
#pragma unroll
    for (int g = 0; g < H_; ++g) {
      float a = pwb[g];
#pragma unroll
      for (int h = 0; h < H_; ++h) a += pv[h] * ww[h * H_ + g];
      P[((long long)g * N_ + q) * N_ + c] = f2bf(a);
    }
  }
}

// ---------------------------------------------------------------------------
// Weight convert+transpose: src f32 [K][Nn] -> dst bf16 [Np][Kp], zero padded.
// ---------------------------------------------------------------------------
__global__ __launch_bounds__(256)
void convert_transpose_kernel(const float* __restrict__ src, unsigned short* __restrict__ dst,
                              int K, int Nn, int Kp, int Np)
{
  long long idx = (long long)blockIdx.x * 256 + threadIdx.x;
  long long total = (long long)Np * Kp;
  if (idx >= total) return;
  const int n = (int)(idx / Kp);
  const int k = (int)(idx % Kp);
  const float v = (k < K && n < Nn) ? src[(long long)k * Nn + n] : 0.f;
  dst[idx] = f2bf(v);
}

__global__ __launch_bounds__(256)
void relu_mul_kernel(unsigned short* __restrict__ wide, const unsigned short* __restrict__ gate,
                     long long total)
{
  long long idx = (long long)blockIdx.x * 256 + threadIdx.x;
  if (idx >= total) return;
  const float w = bf2f(wide[idx]);
  const float g = bf2f(gate[idx]);
  wide[idx] = f2bf(fmaxf(w, 0.f) * g);
}

// ---------------------------------------------------------------------------
extern "C" void kernel_launch(void* const* d_in, const int* in_sizes, int n_in,
                              void* d_out, int out_size, void* d_ws, size_t ws_size,
                              hipStream_t stream)
{
  (void)in_sizes; (void)n_in; (void)out_size; (void)ws_size;
  const float*         x_in    = (const float*)d_in[0];
  const unsigned char* mask    = (const unsigned char*)d_in[1];
  const float* ln1_g  = (const float*)d_in[2];
  const float* ln1_b  = (const float*)d_in[3];
  const float* qkv_w  = (const float*)d_in[4];
  const float* qkv_b  = (const float*)d_in[5];
  const float* projl_w= (const float*)d_in[6];
  const float* projl_b= (const float*)d_in[7];
  const float* projw_w= (const float*)d_in[8];
  const float* projw_b= (const float*)d_in[9];
  const float* proj_w = (const float*)d_in[10];
  const float* proj_b = (const float*)d_in[11];
  const float* ls1    = (const float*)d_in[12];
  const float* ln2_g  = (const float*)d_in[13];
  const float* ln2_b  = (const float*)d_in[14];
  const float* wide_w = (const float*)d_in[15];
  const float* gate_w = (const float*)d_in[16];
  const float* out_w  = (const float*)d_in[17];
  const float* ls2    = (const float*)d_in[18];
  float* out = (float*)d_out;

  // Workspace carve-up (~185 MB with aliasing; scores+probs stay L2-resident)
  char* p = (char*)d_ws;
  size_t off = 0;
  auto take = [&](size_t bytes) -> char* {
    char* r = p + off;
    off += bytes;
    off = (off + 255) & ~(size_t)255;
    return r;
  };
  unsigned short* lnbuf   = (unsigned short*)take(4096ULL * 1024 * 2);        // ln1/ln2 out
  char*           qkvReg  = take(4096ULL * 3072 * 2);                         // 24 MB region
  unsigned short* qkvbuf  = (unsigned short*)qkvReg;                          // fused qkv (dead after split)
  unsigned short* attnOut = (unsigned short*)qkvReg;                          // alias: attn out bf16 (8 MB)
  float*          x1      = (float*)(qkvReg + 4096ULL * 1024 * 2);            // alias: residual f32 (16 MB)
  unsigned short* qbuf  = (unsigned short*)take(4ULL * 16 * 1024 * 64 * 2);
  unsigned short* kbuf  = (unsigned short*)take(4ULL * 16 * 1024 * 64 * 2);
  unsigned short* vtbuf = (unsigned short*)take(4ULL * 16 * 1024 * 64 * 2);
  char*           scReg = take(16ULL * 1024 * 1024 * 4);                      // 64 MB per-batch scores
  float*          scores  = (float*)scReg;
  unsigned short* wideBuf = (unsigned short*)scReg;                           // alias (scores dead post-attn)
  unsigned short* gateBuf = (unsigned short*)(scReg + 4096ULL * HIDP_ * 2);
  unsigned short* pbuf = (unsigned short*)take(16ULL * 1024 * 1024 * 2);      // per-batch bf16 probs
  unsigned short* qkvT = (unsigned short*)take(3072ULL * 1024 * 2);
  unsigned short* projT= (unsigned short*)take(1024ULL * 1024 * 2);
  unsigned short* wideT= (unsigned short*)take((size_t)HIDP_ * 1024 * 2);
  unsigned short* gateT= (unsigned short*)take((size_t)HIDP_ * 1024 * 2);
  unsigned short* outT = (unsigned short*)take(1024ULL * HIDP_ * 2);

  dim3 blk(256);
  auto g1 = [](long long n) { return dim3((unsigned)((n + 255) / 256)); };

  // Weights -> bf16, transposed to [N][K], zero padded to HIDP where needed
  convert_transpose_kernel<<<g1(3072LL * 1024), blk, 0, stream>>>(qkv_w,  qkvT, 1024, 3072, 1024, 3072);
  convert_transpose_kernel<<<g1(1024LL * 1024), blk, 0, stream>>>(proj_w, projT, 1024, 1024, 1024, 1024);
  convert_transpose_kernel<<<g1((long long)HIDP_ * 1024), blk, 0, stream>>>(wide_w, wideT, 1024, HID_, 1024, HIDP_);
  convert_transpose_kernel<<<g1((long long)HIDP_ * 1024), blk, 0, stream>>>(gate_w, gateT, 1024, HID_, 1024, HIDP_);
  convert_transpose_kernel<<<g1(1024LL * HIDP_), blk, 0, stream>>>(out_w, outT, HID_, 1024, HIDP_, 1024);

  // LN1 -> bf16
  layernorm_bf16_kernel<<<dim3(4096), blk, 0, stream>>>(x_in, ln1_g, ln1_b, lnbuf);

  // QKV GEMM (4096 x 3072 x 1024), bf16 out + bias
  gemm_bf16_wmma<1><<<dim3(24, 16, 1), blk, 0, stream>>>(
      lnbuf, 0, 1024, qkvT, 0, 1024, qkvbuf, 0, 3072,
      qkv_b, nullptr, nullptr, 0, 4096, 3072, 1024);

  split_qkv_kernel<<<dim3(16384), blk, 0, stream>>>(qkvbuf, qbuf, kbuf, vtbuf);

  // Attention, batch-sequential so scores (64MB) + probs (32MB) fit in L2
  for (int b = 0; b < B_; ++b) {
    const unsigned short* qb = qbuf  + (long long)b * H_ * N_ * DH_;
    const unsigned short* kb = kbuf  + (long long)b * H_ * N_ * DH_;
    const unsigned short* vb = vtbuf + (long long)b * H_ * DH_ * N_;
    // S_h = Q_h @ K_h^T (per head via gridDim.z): M=N=1024, K=64, f32 out
    gemm_bf16_wmma<0><<<dim3(8, 4, H_), blk, 0, stream>>>(
        qb, (long long)N_ * DH_, DH_, kb, (long long)N_ * DH_, DH_,
        scores, (long long)N_ * N_, N_,
        nullptr, nullptr, nullptr, 0, N_, N_, DH_);
    mix_softmax_kernel<<<dim3(N_), blk, 0, stream>>>(
        scores, mask + (long long)b * N_, projl_w, projl_b, projw_w, projw_b, pbuf);
    // O_h = P_h @ V_h: M=1024, N=64, K=1024, bf16 out packed into (N, H*DH)
    gemm_bf16_wmma<1><<<dim3(1, 4, H_), blk, 0, stream>>>(
        pbuf, (long long)N_ * N_, N_, vb, (long long)DH_ * N_, N_,
        attnOut + (long long)b * N_ * D_, DH_, D_,
        nullptr, nullptr, nullptr, 0, N_, DH_, N_);
  }

  // proj + layer-scale + residual -> x1 (f32)
  gemm_bf16_wmma<2><<<dim3(8, 16, 1), blk, 0, stream>>>(
      attnOut, 0, 1024, projT, 0, 1024, x1, 0, 1024,
      proj_b, ls1, x_in, 1024, 4096, 1024, 1024);

  // LN2 -> bf16
  layernorm_bf16_kernel<<<dim3(4096), blk, 0, stream>>>(x1, ln2_g, ln2_b, lnbuf);

  // wide & gate GEMMs (N padded to 2752; padded cols compute to exactly 0)
  gemm_bf16_wmma<1><<<dim3(22, 16, 1), blk, 0, stream>>>(
      lnbuf, 0, 1024, wideT, 0, 1024, wideBuf, 0, HIDP_,
      nullptr, nullptr, nullptr, 0, 4096, HIDP_, 1024);
  gemm_bf16_wmma<1><<<dim3(22, 16, 1), blk, 0, stream>>>(
      lnbuf, 0, 1024, gateT, 0, 1024, gateBuf, 0, HIDP_,
      nullptr, nullptr, nullptr, 0, 4096, HIDP_, 1024);
  relu_mul_kernel<<<g1(4096LL * HIDP_), blk, 0, stream>>>(wideBuf, gateBuf, 4096LL * HIDP_);

  // out GEMM (K=2752) + layer-scale + residual -> d_out (f32)
  gemm_bf16_wmma<2><<<dim3(8, 16, 1), blk, 0, stream>>>(
      wideBuf, 0, HIDP_, outT, 0, HIDP_, out, 0, 1024,
      nullptr, ls2, x1, 1024, 4096, 1024, HIDP_);
}
